// LSTM_16131897163883
// MI455X (gfx1250) — compile-verified, matrix operand order checked by reading, not run on hardware
//
#include <hip/hip_runtime.h>
#include <cstdint>

typedef unsigned short u16;
typedef unsigned int   u32;
typedef __attribute__((ext_vector_type(16))) __bf16 v16bf;
typedef __attribute__((ext_vector_type(8)))  float  v8f;
typedef __attribute__((ext_vector_type(8)))  u32    v8u;

#define T_STEPS 1024
#define Bsz 256
#define Dsz 128
#define Hsz 512
#define Ksz 640   // D + H
#define NCH 20    // Ksz / 32

__device__ __forceinline__ u16 f2bf(float f) {
    u32 u = __builtin_bit_cast(u32, f);
    u32 r = (u + 0x7FFFu + ((u >> 16) & 1u)) >> 16;
    return (u16)r;
}
__device__ __forceinline__ float bf2f(u16 h) {
    return __builtin_bit_cast(float, (u32)h << 16);
}
__device__ __forceinline__ v16bf mkfrag(uint4 lo, uint4 hi) {
    v8u u = (v8u){lo.x, lo.y, lo.z, lo.w, hi.x, hi.y, hi.z, hi.w};
    return __builtin_bit_cast(v16bf, u);
}

// ---------------------------------------------------------------------------
// Pack [Wx|Wh] for all 4 gates into per-lane WMMA B-fragment order (bf16).
// Bp flat index = ((chunk*2 + half)*2048 + n)*16 + pos, pos 0..15 are the
// 16 bf16 one lane loads for one 16x16x32 WMMA K-chunk.
// K element for (pos,half): k' = (pos&7) + 8*half + 16*(pos>>3)
// ---------------------------------------------------------------------------
__global__ void pack_weights(const float* __restrict__ Wgx, const float* __restrict__ Wgh,
                             const float* __restrict__ Wix, const float* __restrict__ Wih,
                             const float* __restrict__ Wfx, const float* __restrict__ Wfh,
                             const float* __restrict__ Wox, const float* __restrict__ Woh,
                             u16* __restrict__ Bp) {
    int e    = blockIdx.x * 256 + threadIdx.x;      // 0 .. 40*2048*16-1
    int pos  = e & 15;
    int n    = (e >> 4) & 2047;
    int ch   = e >> 15;                             // chunk*2 + half
    int ck   = ch >> 1, half = ch & 1;
    int kp   = (pos & 7) + 8 * half + ((pos >> 3) << 4);
    int k    = ck * 32 + kp;                        // 0..639
    int g    = n >> 9;                              // gate
    int nn   = n & 511;                             // column inside gate
    const float* Wx; const float* Wh;
    switch (g) {
        case 0:  Wx = Wgx; Wh = Wgh; break;
        case 1:  Wx = Wix; Wh = Wih; break;
        case 2:  Wx = Wfx; Wh = Wfh; break;
        default: Wx = Wox; Wh = Woh; break;
    }
    float v = (k < Dsz) ? Wx[(size_t)nn * Dsz + k]
                        : Wh[(size_t)nn * Hsz + (k - Dsz)];
    Bp[e] = f2bf(v);
}

__global__ void init_state(u16* __restrict__ h0, float* __restrict__ c) {
    int i = blockIdx.x * 256 + threadIdx.x;         // 0 .. 131071
    h0[i] = 0;
    c[i]  = 0.0f;
}

// ---------------------------------------------------------------------------
// One LSTM timestep, fully fused: GEMM (x_t,h_prev)->4 gate pre-acts via
// v_wmma_f32_16x16x32_bf16, then gate nonlinearities + c/h update in epilogue.
// Grid: 64 blocks x 256 threads (8 waves). Block = 16 batch rows (M tile),
// wave = one 16-wide h slice across all 4 gates. K-loop software-pipelined:
// chunk ck+1 fragment loads issue before chunk ck's four WMMAs.
// ---------------------------------------------------------------------------
__global__ void __launch_bounds__(256)
lstm_step(const float* __restrict__ x, const u16* __restrict__ Bp,
          const float* __restrict__ bg, const float* __restrict__ bi,
          const float* __restrict__ bff, const float* __restrict__ bo,
          const u16* __restrict__ h_in, u16* __restrict__ h_out,
          float* __restrict__ cst, int t) {
    __shared__ __align__(16) u16 As[16][648];       // 16 x 640 bf16, padded rows

    const int tid   = threadIdx.x;
    const int lane  = tid & 31;
    const int wave  = tid >> 5;
    const int mtile = blockIdx.x >> 2;              // 0..15
    const int hgrp  = blockIdx.x & 3;               // 0..3
    const int m0    = mtile * 16;
    const int h0    = (hgrp * 8 + wave) * 16;

    // ---- cooperative A-tile load: cols 0..127 = bf16(x[b,t,:]), 128..639 = h_prev[b,:]
    {
        const int row = tid & 15;
        const int q   = tid >> 4;                   // 0..15
        const int b   = m0 + row;
        const float* xr = x + ((size_t)b * T_STEPS + t) * Dsz + q * 8;
        float4 f0 = *(const float4*)(xr);
        float4 f1 = *(const float4*)(xr + 4);
        u32 w0 = (u32)f2bf(f0.x) | ((u32)f2bf(f0.y) << 16);
        u32 w1 = (u32)f2bf(f0.z) | ((u32)f2bf(f0.w) << 16);
        u32 w2 = (u32)f2bf(f1.x) | ((u32)f2bf(f1.y) << 16);
        u32 w3 = (u32)f2bf(f1.z) | ((u32)f2bf(f1.w) << 16);
        *(uint4*)&As[row][q * 8] = make_uint4(w0, w1, w2, w3);
        const uint4* hr = (const uint4*)(h_in + (size_t)b * Hsz + q * 32);
        uint4* dst = (uint4*)&As[row][Dsz + q * 32];
        dst[0] = hr[0]; dst[1] = hr[1]; dst[2] = hr[2]; dst[3] = hr[3];
    }
    __syncthreads();

    const int half = lane >> 4;
    const int ln   = lane & 15;
    const int j    = h0 + ln;                       // hidden index this lane owns

    v8f acc[4];
    {
        float b0 = bg[j], b1 = bi[j], b2 = bff[j], b3 = bo[j];
        acc[0] = (v8f){b0, b0, b0, b0, b0, b0, b0, b0};
        acc[1] = (v8f){b1, b1, b1, b1, b1, b1, b1, b1};
        acc[2] = (v8f){b2, b2, b2, b2, b2, b2, b2, b2};
        acc[3] = (v8f){b3, b3, b3, b3, b3, b3, b3, b3};
    }

    const u16* rowp  = &As[ln][0] + 8 * half;       // A frag base for this lane
    const u16* bpane = Bp + ((size_t)half * 2048 + j) * 16;  // B frag base

    // ---- software-pipelined K loop ------------------------------------
    uint4 alo, ahi, blo[4], bhi[4];
    {   // prologue: chunk 0
        const u16* ap = rowp;
        alo = *(const uint4*)(ap);
        ahi = *(const uint4*)(ap + 16);
        const u16* bb = bpane;
#pragma unroll
        for (int g = 0; g < 4; ++g) {
            const u16* bp = bb + (size_t)g * Hsz * 16;
            blo[g] = *(const uint4*)(bp);
            bhi[g] = *(const uint4*)(bp + 8);
        }
    }
#pragma unroll 2
    for (int ck = 0; ck < NCH - 1; ++ck) {
        // issue chunk ck+1 loads BEFORE consuming chunk ck fragments
        uint4 nalo, nahi, nblo[4], nbhi[4];
        {
            const u16* ap = rowp + (ck + 1) * 32;
            nalo = *(const uint4*)(ap);
            nahi = *(const uint4*)(ap + 16);
            const u16* bb = bpane + (size_t)(ck + 1) * 2 * 2048 * 16;
#pragma unroll
            for (int g = 0; g < 4; ++g) {
                const u16* bp = bb + (size_t)g * Hsz * 16;
                nblo[g] = *(const uint4*)(bp);
                nbhi[g] = *(const uint4*)(bp + 8);
            }
        }
        v16bf afr = mkfrag(alo, ahi);
#pragma unroll
        for (int g = 0; g < 4; ++g) {
            acc[g] = __builtin_amdgcn_wmma_f32_16x16x32_bf16(
                false, afr, false, mkfrag(blo[g], bhi[g]), (short)0, acc[g],
                false, false);
        }
        alo = nalo; ahi = nahi;
#pragma unroll
        for (int g = 0; g < 4; ++g) { blo[g] = nblo[g]; bhi[g] = nbhi[g]; }
    }
    {   // epilogue chunk NCH-1
        v16bf afr = mkfrag(alo, ahi);
#pragma unroll
        for (int g = 0; g < 4; ++g) {
            acc[g] = __builtin_amdgcn_wmma_f32_16x16x32_bf16(
                false, afr, false, mkfrag(blo[g], bhi[g]), (short)0, acc[g],
                false, false);
        }
    }

    // ---- fused gate epilogue: c = tanh(pre_g)*sig(pre_i) + c*sig(pre_f);
    //      h = tanh(c)*sig(pre_o)
#pragma unroll
    for (int e = 0; e < 8; ++e) {
        int brow = m0 + e + 8 * half;               // C/D layout: VGPR e, half
        float pg = acc[0][e], pi = acc[1][e], pf = acc[2][e], po = acc[3][e];
        float gg = tanhf(pg);
        float ii = 1.0f / (1.0f + __expf(-pi));
        float ff = 1.0f / (1.0f + __expf(-pf));
        float oo = 1.0f / (1.0f + __expf(-po));
        size_t idx = (size_t)brow * Hsz + j;
        float cn = gg * ii + cst[idx] * ff;
        cst[idx] = cn;
        h_out[idx] = f2bf(tanhf(cn) * oo);
    }
}

// ---------------------------------------------------------------------------
// Final tiny projection: out[b,c] = b_p[c] + sum_j h[b,j] * W_ph[c,j]
// ---------------------------------------------------------------------------
__global__ void proj_kernel(const u16* __restrict__ h, const float* __restrict__ Wp,
                            const float* __restrict__ bp, float* __restrict__ out) {
    int tid = blockIdx.x * 256 + threadIdx.x;       // 0..2559
    int b = tid / 10, cc = tid % 10;
    const u16*   hr = h  + (size_t)b * Hsz;
    const float* wr = Wp + (size_t)cc * Hsz;
    float s = bp[cc];
    for (int jj = 0; jj < Hsz; ++jj) s += bf2f(hr[jj]) * wr[jj];
    out[tid] = s;
}

extern "C" void kernel_launch(void* const* d_in, const int* in_sizes, int n_in,
                              void* d_out, int out_size, void* d_ws, size_t ws_size,
                              hipStream_t stream) {
    const float* x   = (const float*)d_in[0];
    const float* Wgx = (const float*)d_in[1];
    const float* Wgh = (const float*)d_in[2];
    const float* bg  = (const float*)d_in[3];
    const float* Wix = (const float*)d_in[4];
    const float* Wih = (const float*)d_in[5];
    const float* bi  = (const float*)d_in[6];
    const float* Wfx = (const float*)d_in[7];
    const float* Wfh = (const float*)d_in[8];
    const float* bff = (const float*)d_in[9];
    const float* Wox = (const float*)d_in[10];
    const float* Woh = (const float*)d_in[11];
    const float* bo  = (const float*)d_in[12];
    const float* Wph = (const float*)d_in[13];
    const float* bp  = (const float*)d_in[14];

    char* ws = (char*)d_ws;
    u16*   Bp   = (u16*)(ws);                       // 640*2048*2   = 2,621,440 B
    u16*   hA   = (u16*)(ws + 2621440);             // 256*512*2    =   262,144 B
    u16*   hB   = (u16*)(ws + 2621440 + 262144);    //                 262,144 B
    float* cbuf = (float*)(ws + 2621440 + 524288);  // 256*512*4    =   524,288 B

    // one-time weight swizzle (amortized over 1024 steps) + state init
    pack_weights<<<5120, 256, 0, stream>>>(Wgx, Wgh, Wix, Wih, Wfx, Wfh, Wox, Woh, Bp);
    init_state<<<512, 256, 0, stream>>>(hA, cbuf);

    // sequential recurrence: one fused GEMM+gate kernel per timestep,
    // ping-pong h buffers (stream order is the cross-step barrier)
    for (int t = 0; t < T_STEPS; ++t) {
        const u16* hin = (t & 1) ? hB : hA;
        u16*      hout = (t & 1) ? hA : hB;
        lstm_step<<<64, 256, 0, stream>>>(x, Bp, bg, bi, bff, bo, hin, hout, cbuf, t);
    }
    // t=1023 wrote hA
    proj_kernel<<<10, 256, 0, stream>>>(hA, Wph, bp, (float*)d_out);
}
